// ClassificationLSTM_63367947485228
// MI455X (gfx1250) — matrix operand order repaired and checked by reading, not compiled, not run on hardware
//
#include <hip/hip_runtime.h>
#include <hip/hip_bf16.h>

#define BATCH 256
#define TSTEPS 2048
#define HID 128
#define GDIM 512 // 4*HID

#if defined(__has_builtin)
#if __has_builtin(__builtin_amdgcn_global_load_async_to_lds_b128) && \
    __has_builtin(__builtin_amdgcn_s_wait_asynccnt)
#define USE_ASYNC 1
#endif
#endif
#ifndef USE_ASYNC
#define USE_ASYNC 0
#endif

typedef __bf16 bf16;
typedef __attribute__((ext_vector_type(16))) __bf16 v16bf;
typedef __attribute__((ext_vector_type(8)))  float  v8f;

__device__ __forceinline__ v8f wmma_bf16(v16bf a, v16bf b, v8f c) {
  return __builtin_amdgcn_wmma_f32_16x16x32_bf16(false, a, false, b, (short)0, c, false, false);
}

__device__ __forceinline__ float sigm(float x) { return 1.0f / (1.0f + __expf(-x)); }

#if USE_ASYNC
// Exact pointee type from the builtin's signature: int __attribute__((vector_size(16)))
typedef int v4i_t __attribute__((vector_size(16)));
typedef __attribute__((address_space(1))) v4i_t glob_v4i;
typedef __attribute__((address_space(3))) v4i_t lds_v4i;
__device__ __forceinline__ void async_copy_b128(const void* g, void* l) {
  // LDS generic pointers carry the raw LDS offset in the low 32 bits (ISA aperture rule),
  // so an integer round-trip yields the correct as(3)/as(1) pointers.
  __builtin_amdgcn_global_load_async_to_lds_b128(
      (glob_v4i*)(unsigned long long)g,
      (lds_v4i*)(unsigned int)(unsigned long long)l, 0, 0);
}
#endif

// 16x32 bf16 A-fragment from row-major [16][128] bf16 LDS tile (ISA 16-bit A layout).
__device__ __forceinline__ v16bf load_a_frag(const bf16* src, int lane, int koff) {
  union { v16bf v; unsigned u[8]; } a;
  const int M = lane & 15;
  const int base = (lane >> 4) << 2; // (lane/16)*8 halves = *4 dwords
  const unsigned* row = (const unsigned*)(src + M * HID + koff);
#pragma unroll
  for (int v = 0; v < 4; ++v) a.u[v] = row[base + v];
#pragma unroll
  for (int v = 0; v < 4; ++v) a.u[4 + v] = row[8 + base + v];
  return a.v;
}

// B-fragment: each lane's 16 halves contiguous -> two ds_load_b128.
__device__ __forceinline__ v16bf load_b_frag(const bf16* frag, int lane) {
  union { v16bf v; uint4 q[2]; } b;
  const uint4* p = (const uint4*)(frag + lane * 16);
  b.q[0] = p[0];
  b.q[1] = p[1];
  return b.v;
}

__device__ __forceinline__ void wait_progress(const int* p, int need) {
  if (__hip_atomic_load(p, __ATOMIC_ACQUIRE, __HIP_MEMORY_SCOPE_AGENT) >= need) return;
  while (__hip_atomic_load(p, __ATOMIC_ACQUIRE, __HIP_MEMORY_SCOPE_AGENT) < need)
    __builtin_amdgcn_s_sleep(2);
}

// MODE 0: scalar input (layer 0), producer. MODE 1: consumer+producer. MODE 2: consumer, emits hT.
template <int MODE>
__device__ __forceinline__ void run_layer(
    char* smem, int tile,
    const float* __restrict__ x, const bf16* __restrict__ hseq_in,
    const float* __restrict__ Wih, const float* __restrict__ Whh,
    const float* __restrict__ bih, const float* __restrict__ bhh,
    bf16* __restrict__ hseq_out, float* __restrict__ hT_out,
    const int* prog_in, int* prog_out)
{
  constexpr int KC = (MODE == 0) ? 4 : 8; // K = 128 (rec only) or 256 ([in;rec])
  bf16*  wfrag = (bf16*)smem;             // 8*32 frags * 512 halves (fixed layout)
  bf16*  h_cur = wfrag + 8 * 32 * 512;    // [16][128] recurrent state
  bf16*  hstg  = h_cur + 16 * HID;        // [2][16][128] staged h_in (double buffer)
  float* bias  = (float*)(hstg + 2 * 16 * HID);
  float* wx    = bias + GDIM;             // Wih0 column (MODE0)
  float* xs    = wx + GDIM;               // [16][32] x chunk (MODE0)

  const int tid  = threadIdx.x;
  const int lane = tid & 31;
  const int wv   = tid >> 5;
  const int b0   = tile * 16;

  // ---- pack fused weights [Wih;Whh] into WMMA B-fragment order, fp32->bf16 ----
  for (int p = tid; p < KC * 32 * 512; p += 256) {
    int kc = p >> 14, rem = p & 16383, nt = rem >> 9, q = rem & 511;
    int ln = q >> 4, idx = q & 15;
    int Kl = ((idx >> 3) << 4) + ((ln >> 4) << 3) + (idx & 7);
    int N  = (nt << 4) + (ln & 15);
    int Kg = (kc << 5) + Kl;
    float wsrc;
    if (MODE == 0)     wsrc = Whh[N * HID + Kg];
    else if (Kg < HID) wsrc = Wih[N * HID + Kg];
    else               wsrc = Whh[N * HID + (Kg - HID)];
    wfrag[p] = (bf16)wsrc;
  }
  bias[tid]       = bih[tid] + bhh[tid];
  bias[tid + 256] = bih[tid + 256] + bhh[tid + 256];
  if (MODE == 0) { wx[tid] = Wih[tid]; wx[tid + 256] = Wih[tid + 256]; }
#pragma unroll
  for (int k = 0; k < 8; ++k) h_cur[tid * 8 + k] = (bf16)0.0f;

  // ---- prologue: stage h_in(0) into buffer 0 ----
  if (MODE != 0) {
    wait_progress(prog_in, 1);
    const uint4* s0 = (const uint4*)(hseq_in + (size_t)b0 * HID) + tid;
#if USE_ASYNC
    async_copy_b128(s0, (uint4*)hstg + tid);
    __builtin_amdgcn_s_wait_asynccnt(0);
#else
    ((uint4*)hstg)[tid] = *s0;
#endif
  }
  __syncthreads();

  v8f cst = {};
  const int ncol = lane & 15;
  const int roff = (lane >> 4) << 3;
  uint4 stg_reg = {};

  for (int t = 0; t < TSTEPS; ++t) {
    bf16* stg_cur = hstg + (t & 1) * (16 * HID);
    bf16* stg_nxt = hstg + ((t + 1) & 1) * (16 * HID);

    // ---- top: issue next-step staging (overlapped with this step's compute) ----
    if (MODE != 0 && t + 1 < TSTEPS) {
      wait_progress(prog_in, t + 2); // h_in(t+1) globally visible
      const uint4* sn = (const uint4*)(hseq_in + ((size_t)(t + 1) * BATCH + b0) * HID) + tid;
#if USE_ASYNC
      async_copy_b128(sn, (uint4*)stg_nxt + tid);
#else
      stg_reg = *sn;
#endif
      if (t + 2 < TSTEPS) {
        const uint4* pf = (const uint4*)(hseq_in + ((size_t)(t + 2) * BATCH + b0) * HID) + tid;
        __builtin_prefetch((const void*)pf, 0, 1); // global_prefetch_b8
      }
    }
    if (MODE == 0 && (t & 31) == 0) {
#pragma unroll
      for (int h = 0; h < 2; ++h) {
        int e = tid + h * 256, row = e >> 5, col = e & 31;
        xs[row * 32 + col] = x[(size_t)(b0 + row) * TSTEPS + t + col];
      }
      __syncthreads();
    }

    // producer: fence covers h-copies issued at steps <= t-1 (had a full step to land)
    if (MODE < 2 && t > 0) __threadfence();

    // packed write-out of h(t-1): one b128 store per thread
    if (MODE < 2 && t > 0) {
      uint4 hv = ((const uint4*)h_cur)[tid];
      *((uint4*)(hseq_out + ((size_t)(t - 1) * BATCH + b0) * HID) + tid) = hv;
    }

    // ---- read phase: A fragments + accumulator init ----
    v16bf aIn[4], aRec[4];
    if (MODE != 0) {
#pragma unroll
      for (int kc = 0; kc < 4; ++kc) aIn[kc] = load_a_frag(stg_cur, lane, kc * 32);
    }
#pragma unroll
    for (int kc = 0; kc < 4; ++kc) aRec[kc] = load_a_frag(h_cur, lane, kc * 32);

    v8f acc[4];
#pragma unroll
    for (int gi = 0; gi < 4; ++gi) {
      int n = (wv + gi * 8) * 16 + ncol;
      float bv = bias[n];
      if (MODE == 0) {
        float wxn = wx[n];
#pragma unroll
        for (int r = 0; r < 8; ++r) acc[gi][r] = bv + xs[(r + roff) * 32 + (t & 31)] * wxn;
      } else {
#pragma unroll
        for (int r = 0; r < 8; ++r) acc[gi][r] = bv;
      }
    }
    __syncthreads(); // (B): all LDS reads done; safe to overwrite h_cur / staging

    // publish progress = t-1 (h(0..t-2) fenced + visible)
    if (MODE < 2 && t > 1 && tid == 0)
      __hip_atomic_store(prog_out, t - 1, __ATOMIC_RELEASE, __HIP_MEMORY_SCOPE_AGENT);

    // ---- 16x512 gate GEMM, K = KC*32, weights resident in LDS ----
#pragma unroll
    for (int gi = 0; gi < 4; ++gi) {
      const int nt = wv + gi * 8; // wave owns matching i/f/g/o column groups
      if (MODE != 0) {
#pragma unroll
        for (int kc = 0; kc < 4; ++kc)
          acc[gi] = wmma_bf16(aIn[kc], load_b_frag(wfrag + (kc * 32 + nt) * 512, lane), acc[gi]);
      }
#pragma unroll
      for (int kc = 0; kc < 4; ++kc) {
        int kk = (MODE == 0) ? kc : kc + 4;
        acc[gi] = wmma_bf16(aRec[kc], load_b_frag(wfrag + (kk * 32 + nt) * 512, lane), acc[gi]);
      }
    }

    // ---- nonlinearities + cell update (wave-local columns) ----
    const int n = wv * 16 + ncol;
#pragma unroll
    for (int r = 0; r < 8; ++r) {
      float ig = sigm(acc[0][r]);
      float fg = sigm(acc[1][r]);
      float gg = tanhf(acc[2][r]);
      float og = sigm(acc[3][r]);
      float c  = fg * cst[r] + ig * gg;
      cst[r]   = c;
      h_cur[(r + roff) * HID + n] = (bf16)(og * tanhf(c));
    }

    // commit next-step staging to LDS (global latency overlapped with WMMAs above)
#if USE_ASYNC
    if (MODE != 0 && t + 1 < TSTEPS) __builtin_amdgcn_s_wait_asynccnt(0);
#else
    if (MODE != 0 && t + 1 < TSTEPS) ((uint4*)stg_nxt)[tid] = stg_reg;
#endif
    __syncthreads(); // (A): h_cur(t) + staged h_in(t+1) ready
  }

  // ---- epilogue ----
  if (MODE < 2) {
    uint4 hv = ((const uint4*)h_cur)[tid];
    *((uint4*)(hseq_out + ((size_t)(TSTEPS - 1) * BATCH + b0) * HID) + tid) = hv;
    __threadfence();
    __syncthreads();
    if (tid == 0)
      __hip_atomic_store(prog_out, TSTEPS, __ATOMIC_RELEASE, __HIP_MEMORY_SCOPE_AGENT);
  } else {
#pragma unroll
    for (int k = 0; k < 8; ++k) {
      int e = tid * 8 + k;
      hT_out[(size_t)b0 * HID + e] = (float)h_cur[e]; // rows contiguous: (b0+row)*H+col
    }
  }
}

__global__ __launch_bounds__(256) void lstm_pipeline_kernel(
    const float* x,
    const float* Wih0, const float* Whh0, const float* bih0, const float* bhh0,
    const float* Wih1, const float* Whh1, const float* bih1, const float* bhh1,
    const float* Wih2, const float* Whh2, const float* bih2, const float* bhh2,
    bf16* h1, bf16* h2, float* hT, int* prog)
{
  extern __shared__ char smem[];
  const int layer = blockIdx.x >> 4;
  const int tile  = blockIdx.x & 15;
  if (layer == 0) {
    run_layer<0>(smem, tile, x, nullptr, Wih0, Whh0, bih0, bhh0, h1, nullptr,
                 nullptr, prog + tile);
  } else if (layer == 1) {
    run_layer<1>(smem, tile, nullptr, h1, Wih1, Whh1, bih1, bhh1, h2, nullptr,
                 prog + tile, prog + 16 + tile);
  } else {
    run_layer<2>(smem, tile, nullptr, h2, Wih2, Whh2, bih2, bhh2, nullptr, hT,
                 prog + 16 + tile, nullptr);
  }
}

// FC head: out = relu(hT @ fc1_w^T + b1) @ fc2_w^T + b2 ; (256,128)->(256,5)
__global__ __launch_bounds__(64) void fc_head_kernel(
    const float* __restrict__ hT, const float* __restrict__ w1, const float* __restrict__ b1,
    const float* __restrict__ w2, const float* __restrict__ b2, float* __restrict__ out)
{
  __shared__ float hrow[HID];
  __shared__ float z[64];
  const int b = blockIdx.x, j = threadIdx.x;
  hrow[j]      = hT[b * HID + j];
  hrow[j + 64] = hT[b * HID + 64 + j];
  __syncthreads();
  float s = b1[j];
#pragma unroll 4
  for (int k = 0; k < HID; ++k) s += hrow[k] * w1[j * HID + k];
  z[j] = fmaxf(s, 0.0f);
  __syncthreads();
  if (j < 5) {
    float s2 = b2[j];
#pragma unroll 4
    for (int k = 0; k < 64; ++k) s2 += z[k] * w2[j * 64 + k];
    out[b * 5 + j] = s2;
  }
}

extern "C" void kernel_launch(void* const* d_in, const int* in_sizes, int n_in,
                              void* d_out, int out_size, void* d_ws, size_t ws_size,
                              hipStream_t stream) {
  const float* x     = (const float*)d_in[0];
  const float* Wih0  = (const float*)d_in[1];
  const float* Whh0  = (const float*)d_in[2];
  const float* bih0  = (const float*)d_in[3];
  const float* bhh0  = (const float*)d_in[4];
  const float* Wih1  = (const float*)d_in[5];
  const float* Whh1  = (const float*)d_in[6];
  const float* bih1  = (const float*)d_in[7];
  const float* bhh1  = (const float*)d_in[8];
  const float* Wih2  = (const float*)d_in[9];
  const float* Whh2  = (const float*)d_in[10];
  const float* bih2  = (const float*)d_in[11];
  const float* bhh2  = (const float*)d_in[12];
  const float* fc1_w = (const float*)d_in[13];
  const float* fc1_b = (const float*)d_in[14];
  const float* fc2_w = (const float*)d_in[15];
  const float* fc2_b = (const float*)d_in[16];

  const size_t seq_bytes = (size_t)TSTEPS * BATCH * HID * sizeof(bf16); // 128 MB
  bf16*  h1   = (bf16*)d_ws;
  bf16*  h2   = (bf16*)((char*)d_ws + seq_bytes);
  float* hT   = (float*)((char*)d_ws + 2 * seq_bytes);
  int*   prog = (int*)((char*)d_ws + 2 * seq_bytes + (size_t)BATCH * HID * sizeof(float));

  (void)hipMemsetAsync(prog, 0, 32 * sizeof(int), stream);

  // LDS: weights 262144 + h_cur 4096 + staging 8192 + (bias/wx/xs) 6144 = 280576 B (< 320 KB/WGP)
  const size_t lds_bytes =
      (size_t)(8 * 32 * 512 + 16 * HID + 2 * 16 * HID) * 2 + 3 * GDIM * 4;

  lstm_pipeline_kernel<<<dim3(48), dim3(256), lds_bytes, stream>>>(
      x, Wih0, Whh0, bih0, bhh0, Wih1, Whh1, bih1, bhh1, Wih2, Whh2, bih2, bhh2,
      h1, h2, hT, prog);
  fc_head_kernel<<<dim3(BATCH), dim3(64), 0, stream>>>(hT, fc1_w, fc1_b, fc2_w, fc2_b,
                                                       (float*)d_out);
}
